// SimpleLMWithGDN_44263932953376
// MI455X (gfx1250) — matrix operand order repaired and checked by reading, not compiled, not run on hardware
//
#include <hip/hip_runtime.h>
#include <hip/hip_bf16.h>
#include <math.h>

// ---------------- problem constants (match reference) ----------------
#define BB 2
#define TT 1024
#define MT (BB * TT)          // 2048 rows of activations
#define VV 32000
#define HH 2048
#define HK 16
#define HV 32
#define DK 128
#define DV 128
#define QKDIM (HK * DK)       // 2048
#define VDIM  (HV * DV)       // 4096
#define CONV_DIM (2 * QKDIM + VDIM)   // 8192
#define QKVZ_N (2 * QKDIM + 2 * VDIM) // 12288
#define KCONV 4
#define EPS 1e-6f

typedef __attribute__((ext_vector_type(16))) __bf16 v16bf;
typedef __attribute__((ext_vector_type(8)))  float  v8f;

__device__ __forceinline__ float sigmoidf_(float x) { return 1.0f / (1.0f + __expf(-x)); }
__device__ __forceinline__ float siluf_(float x)    { return x * sigmoidf_(x); }

// ---- CDNA5 async global->LDS copy (ASYNCcnt-tracked, §15.18 op 98) ----
__device__ __forceinline__ void async_copy_b128(const float* g, unsigned ldsoff) {
  asm volatile("global_load_async_to_lds_b128 %0, %1, off"
               :: "v"(ldsoff), "v"(g) : "memory");
}

#ifdef __has_builtin
#if __has_builtin(__builtin_amdgcn_s_wait_asynccnt)
#define WAIT_ASYNC(n) __builtin_amdgcn_s_wait_asynccnt(n)
#endif
#endif
#ifndef WAIT_ASYNC
#define WAIT_ASYNC(n) asm volatile("s_wait_asynccnt %0" :: "i"(n) : "memory")
#endif

// ---------------------------------------------------------------------
// 1) Embedding gather: x[bt, :] = table[ids[bt], :]
// ---------------------------------------------------------------------
__global__ void embed_gather(const int* __restrict__ ids,
                             const float* __restrict__ table,
                             float* __restrict__ x) {
  const int row = blockIdx.x;
  const int id  = ids[row];
  const float4* src = (const float4*)(table + (size_t)id * HH);
  float4*       dst = (float4*)(x + (size_t)row * HH);
  for (int j = threadIdx.x; j < HH / 4; j += blockDim.x) dst[j] = src[j];
}

// ---------------------------------------------------------------------
// 2a) LDS-staged double-buffered bf16 WMMA GEMM.
//     C[M,N] = A[M,Kd] * Bw[Kd,N]   (f32 in memory, bf16 into matrix pipes)
//     block = 256 threads = 8 waves; block tile = 32(M) x 128(N), K-step 32.
//     A/B tiles staged via global_load_async_to_lds_b128, double buffered.
//     Requires M%32==0, N%128==0, Kd%32==0.
// ---------------------------------------------------------------------
#define ASTR 36    // padded LDS row stride (floats) for 32-wide A rows
#define BSTR 132   // padded LDS row stride (floats) for 128-wide B rows

__global__ void __launch_bounds__(256)
gemm_bf16_wmma_lds(const float* __restrict__ A, const float* __restrict__ Bw,
                   float* __restrict__ C, int M, int N, int Kd) {
  __shared__ float sA[2][32 * ASTR];
  __shared__ float sB[2][32 * BSTR];

  const int tid  = threadIdx.x;
  const int lane = tid & 31;
  const int wave = tid >> 5;                 // 0..7
  const int m0 = blockIdx.y * 32;
  const int n0 = blockIdx.x * 128;

  // staging coordinates (per thread, one b128 lane-copy each)
  const int aRow = tid >> 3;                 // 0..31
  const int aQ   = (tid & 7) * 4;            // 0..28
  const int bQ   = (tid & 31) * 4;           // 0..124
  const int bR   = tid >> 5;                 // 0..7  (+8*i covers 32 K-rows)

  const int half = lane >> 4;                // 0 or 1
  const int r    = lane & 15;

  v8f acc0 = {}, acc1 = {};
  const int nk = Kd / 32;

  auto issue = [&](int buf, int k0) {
    // A tile: 32 rows x 32 K (f32)
    async_copy_b128(A + (size_t)(m0 + aRow) * Kd + k0 + aQ,
                    (unsigned)(uintptr_t)&sA[buf][aRow * ASTR + aQ]);
    // B tile: 32 K-rows x 128 N (f32)
#pragma unroll
    for (int i = 0; i < 4; ++i) {
      const int row = bR + i * 8;
      async_copy_b128(Bw + (size_t)(k0 + row) * N + n0 + bQ,
                      (unsigned)(uintptr_t)&sB[buf][row * BSTR + bQ]);
    }
  };

  issue(0, 0);
  for (int kt = 0; kt < nk; ++kt) {
    const int cur = kt & 1;
    if (kt + 1 < nk) {
      issue(cur ^ 1, (kt + 1) * 32);
      WAIT_ASYNC(5);                 // 5 outstanding from next tile; current done
    } else {
      WAIT_ASYNC(0);
    }
    __syncthreads();                 // all waves' async data visible

    // ---- B fragment: j<8 -> K=half*8+j ; j>=8 -> +16, column = wave*16 + r ----
    v16bf bf;
#pragma unroll
    for (int j = 0; j < 16; ++j)
      bf[j] = (__bf16)sB[cur][(half * 8 + (j < 8 ? j : j + 8)) * BSTR + wave * 16 + r];

    // ---- A fragments (16-bit A 16x32 layout): j<8 -> K=half*8+j ; j>=8 -> +16 ----
    const float* a0p = &sA[cur][(r)      * ASTR + half * 8];
    const float* a1p = &sA[cur][(16 + r) * ASTR + half * 8];
    v16bf af0, af1;
#pragma unroll
    for (int j = 0; j < 8; ++j) {
      af0[j] = (__bf16)a0p[j];  af0[j + 8] = (__bf16)a0p[16 + j];
      af1[j] = (__bf16)a1p[j];  af1[j + 8] = (__bf16)a1p[16 + j];
    }

    acc0 = __builtin_amdgcn_wmma_f32_16x16x32_bf16(false, af0, false, bf,
                                                   (short)0, acc0, false, false);
    acc1 = __builtin_amdgcn_wmma_f32_16x16x32_bf16(false, af1, false, bf,
                                                   (short)0, acc1, false, false);
    __syncthreads();                 // everyone done reading before buffer reuse
  }

  // ---- C/D layout: VGPR v -> row (half*8 + v), col r ----
  float* cp0 = C + (size_t)(m0 + half * 8)      * N + n0 + wave * 16 + r;
  float* cp1 = C + (size_t)(m0 + 16 + half * 8) * N + n0 + wave * 16 + r;
#pragma unroll
  for (int v = 0; v < 8; ++v) {
    cp0[(size_t)v * N] = acc0[v];
    cp1[(size_t)v * N] = acc1[v];
  }
}

// ---------------------------------------------------------------------
// 2b) Direct-load WMMA GEMM for small N (the [2048 x 64] "ba" projection).
// ---------------------------------------------------------------------
__global__ void __launch_bounds__(256)
gemm_bf16_wmma_direct(const float* __restrict__ A, const float* __restrict__ Bw,
                      float* __restrict__ C, int M, int N, int Kd) {
  const int lane = threadIdx.x & 31;
  const int wave = threadIdx.x >> 5;
  const int m0 = blockIdx.y * 16;
  const int n0 = blockIdx.x * 128 + wave * 16;
  if (n0 >= N) return;                      // wave-uniform exit

  const int half = lane >> 4;
  const int r    = lane & 15;
  const int arow = m0 + r;
  const int bcol = n0 + r;

  v8f acc = {};
  for (int k0 = 0; k0 < Kd; k0 += 32) {
    const float* ap0 = A + (size_t)arow * Kd + k0 + half * 8;
    const float* ap1 = ap0 + 16;
    float4 a0 = *(const float4*)(ap0);
    float4 a1 = *(const float4*)(ap0 + 4);
    float4 a2 = *(const float4*)(ap1);
    float4 a3 = *(const float4*)(ap1 + 4);
    v16bf af;
    af[0]  = (__bf16)a0.x;  af[1]  = (__bf16)a0.y;  af[2]  = (__bf16)a0.z;  af[3]  = (__bf16)a0.w;
    af[4]  = (__bf16)a1.x;  af[5]  = (__bf16)a1.y;  af[6]  = (__bf16)a1.z;  af[7]  = (__bf16)a1.w;
    af[8]  = (__bf16)a2.x;  af[9]  = (__bf16)a2.y;  af[10] = (__bf16)a2.z;  af[11] = (__bf16)a2.w;
    af[12] = (__bf16)a3.x;  af[13] = (__bf16)a3.y;  af[14] = (__bf16)a3.z;  af[15] = (__bf16)a3.w;

    const float* bp = Bw + (size_t)(k0 + half * 8) * N + bcol;
    v16bf bf;
#pragma unroll
    for (int j = 0; j < 16; ++j) bf[j] = (__bf16)bp[(size_t)(j < 8 ? j : j + 8) * N];

    acc = __builtin_amdgcn_wmma_f32_16x16x32_bf16(false, af, false, bf,
                                                  (short)0, acc, false, false);
  }

  float* cp = C + (size_t)(m0 + half * 8) * N + bcol;
#pragma unroll
  for (int v = 0; v < 8; ++v) cp[(size_t)v * N] = acc[v];
}

// ---------------------------------------------------------------------
// 3) causal depthwise conv1d (K=4) over first 8192 cols of qkvz, + bias + SiLU
// ---------------------------------------------------------------------
__global__ void conv_silu(const float* __restrict__ qkvz,
                          const float* __restrict__ conv_w,
                          const float* __restrict__ conv_b,
                          float* __restrict__ qkv) {
  const int c  = blockIdx.x * blockDim.x + threadIdx.x;   // channel 0..8191
  const int bt = blockIdx.y;                              // 0..MT-1
  const int b  = bt / TT, t = bt % TT;
  float acc = conv_b[c];
#pragma unroll
  for (int j = 0; j < KCONV; ++j) {
    const int tt = t - (KCONV - 1) + j;
    if (tt >= 0)
      acc = fmaf(qkvz[((size_t)(b * TT + tt)) * QKVZ_N + c], conv_w[c * KCONV + j], acc);
  }
  qkv[(size_t)bt * CONV_DIM + c] = siluf_(acc);
}

// ---------------------------------------------------------------------
// 4) per-head l2norm of q (scaled by DK^-0.5) and k, in place
// ---------------------------------------------------------------------
__global__ void l2norm_qk(float* __restrict__ qkv) {
  const int bt = blockIdx.x;
  const int hh = blockIdx.y;                 // [0,HK) = q heads, [HK,2HK) = k heads
  const int off = (hh < HK) ? hh * DK : QKDIM + (hh - HK) * DK;
  float* p = qkv + (size_t)bt * CONV_DIM + off;
  const int tid = threadIdx.x;
  float v = p[tid];
  __shared__ float red[128];
  red[tid] = v * v;
  __syncthreads();
  for (int s = 64; s > 0; s >>= 1) {
    if (tid < s) red[tid] += red[tid + s];
    __syncthreads();
  }
  float inv = rsqrtf(red[0] + EPS);
  if (hh < HK) inv *= 0.08838834764831845f;  // DK^-0.5
  p[tid] = v * inv;
}

// ---------------------------------------------------------------------
// 5) beta = sigmoid(b), g = -exp(A_log)*softplus(a + dt_bias)
// ---------------------------------------------------------------------
__global__ void beta_g(const float* __restrict__ ba,
                       const float* __restrict__ A_log,
                       const float* __restrict__ dt_bias,
                       float* __restrict__ beta, float* __restrict__ g) {
  const int i  = blockIdx.x * blockDim.x + threadIdx.x;   // 0..MT*HV-1
  const int h  = i % HV;
  const int bt = i / HV;
  const float bb = ba[(size_t)bt * (2 * HV) + h];
  const float aa = ba[(size_t)bt * (2 * HV) + HV + h];
  beta[i] = sigmoidf_(bb);
  const float xx = aa + dt_bias[h];
  const float sp = (xx > 20.0f) ? xx : log1pf(__expf(xx));
  g[i] = -__expf(A_log[h]) * sp;
}

// ---------------------------------------------------------------------
// 6) gated delta-rule scan. One block per (b, v-head); 128 threads, lane v
//    owns state column S[:,v] (128 floats) in registers (fully unrolled).
// ---------------------------------------------------------------------
__global__ void __launch_bounds__(128)
gdn_scan(const float* __restrict__ qkv, const float* __restrict__ beta,
         const float* __restrict__ g, float* __restrict__ o) {
  const int bh = blockIdx.x;            // 0..B*HV-1
  const int b  = bh / HV, h = bh % HV;
  const int kh = h >> 1;                // GQA: rep = HV/HK = 2
  const int v  = threadIdx.x;           // 0..127 = output/state column

  float Scol[DK];
#pragma unroll
  for (int k = 0; k < DK; ++k) Scol[k] = 0.0f;

  __shared__ float qt[DK], kt[DK], vt[DV];

  for (int t = 0; t < TT; ++t) {
    const size_t bt = (size_t)(b * TT + t);
    const float* row = qkv + bt * CONV_DIM;
    __syncthreads();                              // previous step done with vectors
    qt[v] = row[kh * DK + v];
    kt[v] = row[QKDIM + kh * DK + v];
    vt[v] = row[2 * QKDIM + h * DV + v];
    const float gt = g[bt * HV + h];
    const float be = beta[bt * HV + h];
    __syncthreads();

    const float eg = __expf(gt);
    float acc = 0.0f;
#pragma unroll
    for (int k = 0; k < DK; ++k) {                // decay + kv = k . S
      const float s = Scol[k] * eg;
      acc = fmaf(kt[k], s, acc);
      Scol[k] = s;
    }
    const float delta = (vt[v] - acc) * be;
    float ov = 0.0f;
#pragma unroll
    for (int k = 0; k < DK; ++k) {                // rank-1 update + o = q . S
      const float s = fmaf(kt[k], delta, Scol[k]);
      Scol[k] = s;
      ov = fmaf(qt[k], s, ov);
    }
    o[bt * VDIM + h * DV + v] = ov;
  }
}

// ---------------------------------------------------------------------
// 7) gated RMSNorm: o = rmsnorm(o * silu(z)) * norm_w   (per head, DV=128)
// ---------------------------------------------------------------------
__global__ void gated_rmsnorm(float* __restrict__ o,
                              const float* __restrict__ qkvz,
                              const float* __restrict__ norm_w) {
  const int bt = blockIdx.x, h = blockIdx.y, v = threadIdx.x;
  const float z = qkvz[(size_t)bt * QKVZ_N + (2 * QKDIM + VDIM) + h * DV + v];
  float* op = o + (size_t)bt * VDIM + h * DV;
  const float xx = op[v] * siluf_(z);
  __shared__ float red[128];
  red[v] = xx * xx;
  __syncthreads();
  for (int s = 64; s > 0; s >>= 1) {
    if (v < s) red[v] += red[v + s];
    __syncthreads();
  }
  const float mean = red[0] * (1.0f / (float)DV);
  op[v] = xx * rsqrtf(mean + EPS) * norm_w[v];
}

// ---------------------------------------------------------------------
extern "C" void kernel_launch(void* const* d_in, const int* in_sizes, int n_in,
                              void* d_out, int out_size, void* d_ws, size_t ws_size,
                              hipStream_t stream) {
  const int*   ids     = (const int*)  d_in[0];
  const float* table   = (const float*)d_in[1];
  const float* W_qkvz  = (const float*)d_in[2];
  const float* W_ba    = (const float*)d_in[3];
  const float* conv_w  = (const float*)d_in[4];
  const float* conv_b  = (const float*)d_in[5];
  const float* A_log   = (const float*)d_in[6];
  const float* dt_bias = (const float*)d_in[7];
  const float* norm_w  = (const float*)d_in[8];
  const float* W_out   = (const float*)d_in[9];
  const float* W_lm    = (const float*)d_in[10];
  float* logits = (float*)d_out;

  float* ws   = (float*)d_ws;
  float* x    = ws;  ws += (size_t)MT * HH;        // [2048, 2048]
  float* qkvz = ws;  ws += (size_t)MT * QKVZ_N;    // [2048, 12288]
  float* ba   = ws;  ws += (size_t)MT * 2 * HV;    // [2048, 64]
  float* qkv  = ws;  ws += (size_t)MT * CONV_DIM;  // [2048, 8192]
  float* beta = ws;  ws += (size_t)MT * HV;        // [2048, 32]
  float* g    = ws;  ws += (size_t)MT * HV;        // [2048, 32]
  float* o    = ws;  ws += (size_t)MT * VDIM;      // [2048, 4096]
  float* h2   = ws;  ws += (size_t)MT * HH;        // [2048, 2048]

  embed_gather<<<MT, 256, 0, stream>>>(ids, table, x);

  gemm_bf16_wmma_lds<<<dim3(QKVZ_N / 128, MT / 32), 256, 0, stream>>>(x, W_qkvz, qkvz, MT, QKVZ_N, HH);
  gemm_bf16_wmma_direct<<<dim3(1, MT / 16), 256, 0, stream>>>(x, W_ba, ba, MT, 2 * HV, HH);

  conv_silu<<<dim3(CONV_DIM / 256, MT), 256, 0, stream>>>(qkvz, conv_w, conv_b, qkv);
  l2norm_qk<<<dim3(MT, 2 * HK), 128, 0, stream>>>(qkv);
  beta_g<<<(MT * HV) / 256, 256, 0, stream>>>(ba, A_log, dt_bias, beta, g);

  gdn_scan<<<BB * HV, 128, 0, stream>>>(qkv, beta, g, o);

  gated_rmsnorm<<<dim3(MT, HV), 128, 0, stream>>>(o, qkvz, norm_w);

  gemm_bf16_wmma_lds<<<dim3(HH / 128, MT / 32), 256, 0, stream>>>(o, W_out, h2, MT, HH, VDIM);
  gemm_bf16_wmma_lds<<<dim3(VV / 128, MT / 32), 256, 0, stream>>>(h2, W_lm, logits, MT, VV, HH);
}
